// CrossSelfDecoder_68341519614430
// MI455X (gfx1250) — matrix operand order repaired and checked by reading, not compiled, not run on hardware
//
#include <hip/hip_runtime.h>
#include <math.h>

// ---------------------------------------------------------------------------
// Types / helpers
// ---------------------------------------------------------------------------
typedef unsigned short u16;
typedef __attribute__((ext_vector_type(16))) __bf16 v16bf;
typedef __attribute__((ext_vector_type(8)))  float  v8f;
typedef __attribute__((ext_vector_type(4)))  int    i32x4;

// gfx1250 async global->LDS path (ASYNCcnt), guarded so host pass / older
// toolchains fall back to load + ds_store.
#if defined(__has_builtin)
#if __has_builtin(__builtin_amdgcn_global_load_async_to_lds_b128) && \
    __has_builtin(__builtin_amdgcn_s_wait_asynccnt)
#define USE_ASYNC_LDS 1
#endif
#endif

static __device__ __forceinline__ u16 f2bf(float f) {
  union { float f; unsigned int u; } x; x.f = f;
  unsigned int r = x.u + 0x7FFFu + ((x.u >> 16) & 1u);   // round-to-nearest-even
  return (u16)(r >> 16);
}

// Build a 16-element bf16 fragment from two 16-byte chunks.
static __device__ __forceinline__ v16bf frag2(const u16* p0, const u16* p1) {
  union { v16bf v; uint4 q[2]; } u;
  u.q[0] = *reinterpret_cast<const uint4*>(p0);
  u.q[1] = *reinterpret_cast<const uint4*>(p1);
  return u.v;
}

static __device__ __forceinline__ v8f wmma_bf16(v16bf a, v16bf b, v8f c) {
  // D = A(16x32) * B(32x16) + C(16x16), fp32 accumulate
  return __builtin_amdgcn_wmma_f32_16x16x32_bf16(false, a, false, b, (short)0, c,
                                                 false, false);
}

static __device__ __forceinline__ v8f v8zero() {
  v8f z;
#pragma unroll
  for (int i = 0; i < 8; ++i) z[i] = 0.0f;
  return z;
}

// reductions across a 16-lane half of a wave32
static __device__ __forceinline__ float rmax16(float v) {
#pragma unroll
  for (int m = 8; m >= 1; m >>= 1) v = fmaxf(v, __shfl_xor(v, m, 16));
  return v;
}
static __device__ __forceinline__ float rsum16(float v) {
#pragma unroll
  for (int m = 8; m >= 1; m >>= 1) v += __shfl_xor(v, m, 16);
  return v;
}

#if defined(USE_ASYNC_LDS)
static __device__ __forceinline__ void async_ld_b128(const void* g, void* l) {
  __builtin_amdgcn_global_load_async_to_lds_b128(
      (__attribute__((address_space(1))) i32x4*)const_cast<void*>(g),
      (__attribute__((address_space(3))) i32x4*)l, 0, 0);
}
#endif

// ---------------------------------------------------------------------------
// fp32 -> bf16 convert (vectorized by 4)
// ---------------------------------------------------------------------------
__global__ __launch_bounds__(256) void cvt_bf16_kernel(const float* __restrict__ in,
                                                       u16* __restrict__ out, int n4) {
  int i = blockIdx.x * 256 + threadIdx.x;
  if (i >= n4) return;
  const float4 v = reinterpret_cast<const float4*>(in)[i];
  ushort4 o;
  o.x = f2bf(v.x); o.y = f2bf(v.y); o.z = f2bf(v.z); o.w = f2bf(v.w);
  reinterpret_cast<ushort4*>(out)[i] = o;
}

// fp32 weight [K][Nw] -> bf16 transposed [Nw][K] (B operand, K contiguous)
__global__ __launch_bounds__(256) void wtrans_bf16_kernel(const float* __restrict__ W,
                                                          u16* __restrict__ WT,
                                                          int K, int Nw) {
  size_t idx = (size_t)blockIdx.x * 256 + threadIdx.x;
  size_t total = (size_t)K * Nw;
  if (idx >= total) return;
  int n = (int)(idx / K), k = (int)(idx % K);
  WT[idx] = f2bf(W[(size_t)k * Nw + n]);
}

// ---------------------------------------------------------------------------
// bf16 WMMA GEMM: C[M,N] = A[M,K] * BT[N,K]^T + bias, epilogue: gelu/residual,
// optional fp32 and/or bf16 outputs. Block tile 128x128, wave tile 64x32.
// 8 v_wmma per wave per 32-wide K step.
// ---------------------------------------------------------------------------
__global__ __launch_bounds__(256) void gemm_bf16_kernel(
    const u16* __restrict__ A, const u16* __restrict__ BT,
    const float* __restrict__ bias, const float* __restrict__ res,
    float* __restrict__ outf, u16* __restrict__ outb,
    int M, int Nn, int K, int gelu) {
  __shared__ __align__(16) u16 As[128 * 32];
  __shared__ __align__(16) u16 Bs[128 * 32];

  const int t    = threadIdx.x;
  const int lane = t & 31;
  const int wave = t >> 5;
  const int hi   = lane >> 4;
  const int lo   = lane & 15;
  const int kb   = hi * 8;
  const int m0   = blockIdx.y * 128;
  const int n0   = blockIdx.x * 128;
  const int wm   = (wave >> 2) * 64;   // 2 wave rows
  const int wn   = (wave & 3) * 32;    // 4 wave cols

  v8f acc[4][2];
#pragma unroll
  for (int i = 0; i < 4; ++i)
#pragma unroll
    for (int j = 0; j < 2; ++j) acc[i][j] = v8zero();

  const int srow = t >> 1, scol = (t & 1) * 16;  // 256 thr * 16 elems = 128x32

  for (int k0 = 0; k0 < K; k0 += 32) {
    __syncthreads();
    {
      const uint4* gA = reinterpret_cast<const uint4*>(A + (size_t)(m0 + srow) * K + k0 + scol);
      uint4* sA = reinterpret_cast<uint4*>(&As[srow * 32 + scol]);
      sA[0] = gA[0]; sA[1] = gA[1];
      const uint4* gB = reinterpret_cast<const uint4*>(BT + (size_t)(n0 + srow) * K + k0 + scol);
      uint4* sB = reinterpret_cast<uint4*>(&Bs[srow * 32 + scol]);
      sB[0] = gB[0]; sB[1] = gB[1];
    }
    __syncthreads();

    v16bf a[4], b[2];
#pragma unroll
    for (int i = 0; i < 4; ++i) {
      const u16* p = &As[(wm + i * 16 + lo) * 32 + kb];
      a[i] = frag2(p, p + 16);
    }
#pragma unroll
    for (int j = 0; j < 2; ++j) {
      const u16* p = &Bs[(wn + j * 16 + lo) * 32 + kb];
      b[j] = frag2(p, p + 16);
    }
#pragma unroll
    for (int i = 0; i < 4; ++i)
#pragma unroll
      for (int j = 0; j < 2; ++j) acc[i][j] = wmma_bf16(a[i], b[j], acc[i][j]);
  }

#pragma unroll
  for (int i = 0; i < 4; ++i)
#pragma unroll
    for (int j = 0; j < 2; ++j)
#pragma unroll
      for (int r = 0; r < 8; ++r) {
        int row = m0 + wm + i * 16 + r + hi * 8;
        int col = n0 + wn + j * 16 + lo;
        float v = acc[i][j][r] + bias[col];
        if (gelu) v = 0.5f * v * (1.0f + erff(v * 0.70710678118654752f));
        size_t idx = (size_t)row * Nn + col;
        if (res)  v += res[idx];
        if (outf) outf[idx] = v;
        if (outb) outb[idx] = f2bf(v);
      }
}

// ---------------------------------------------------------------------------
// Flash-style attention, no 1/sqrt(d) scale (faithful to reference).
// Q/K/V bf16 strided; output fp32 contiguous (B,H,N,Dp) == reference's flat
// (B,N,D) reshape without head re-permutation.
// Block = 128 query rows of one (b,h); 4 waves x 32 rows; 16 wmma / 32 keys.
// K-tile staged with gfx1250 async global->LDS when available.
// ---------------------------------------------------------------------------
__global__ __launch_bounds__(128) void attn_kernel(
    const u16* __restrict__ Qp, int qs,
    const u16* __restrict__ Kp, const u16* __restrict__ Vp, int kvs,
    float* __restrict__ Op) {
  constexpr int AN = 1024, AH = 16, ADP = 64;
  __shared__ __align__(16) u16 Ks[32 * 64];        // [key][d]
  __shared__ __align__(16) u16 Vs[64 * 32];        // transposed [d][key]
  __shared__ __align__(16) u16 Ps[4 * 2 * 16 * 32];// per-wave, 2 row tiles

  const int t = threadIdx.x, lane = t & 31, wave = t >> 5;
  const int hi = lane >> 4, lo = lane & 15;
  const int b = blockIdx.z, h = blockIdx.y, q0 = blockIdx.x * 128;
  const int kb = hi * 8;

  // Q fragments: 2 row tiles x 2 K-chunks
  v16bf aq[2][2];
#pragma unroll
  for (int rt = 0; rt < 2; ++rt) {
    const int qrow = q0 + wave * 32 + rt * 16 + lo;
    const u16* qbase = Qp + (size_t)(b * AN + qrow) * qs + h * ADP;
#pragma unroll
    for (int kc = 0; kc < 2; ++kc)
      aq[rt][kc] = frag2(qbase + kc * 32 + kb, qbase + kc * 32 + kb + 16);
  }

  float m[2][8], l[2][8];
  v8f o[2][4];
#pragma unroll
  for (int rt = 0; rt < 2; ++rt) {
#pragma unroll
    for (int r = 0; r < 8; ++r) { m[rt][r] = -INFINITY; l[rt][r] = 0.0f; }
#pragma unroll
    for (int d = 0; d < 4; ++d) o[rt][d] = v8zero();
  }

  u16* Pw = &Ps[wave * 1024];
  const int key = t >> 2, d0 = (t & 3) * 16;  // staging map: 128 thr * 16 elems

  for (int k0 = 0; k0 < AN; k0 += 32) {
    __syncthreads();  // previous iteration's LDS reads done
    {
      const u16* gk = Kp + (size_t)(b * AN + k0 + key) * kvs + h * ADP + d0;
#if defined(USE_ASYNC_LDS)
      async_ld_b128(gk, &Ks[key * 64 + d0]);
      async_ld_b128(gk + 8, &Ks[key * 64 + d0 + 8]);
#else
      const uint4* gK = reinterpret_cast<const uint4*>(gk);
      uint4* sK = reinterpret_cast<uint4*>(&Ks[key * 64 + d0]);
      sK[0] = gK[0]; sK[1] = gK[1];
#endif
      // V staged transposed through registers
      union { uint4 q[2]; u16 u[16]; } vv;
      const u16* gv = Vp + (size_t)(b * AN + k0 + key) * kvs + h * ADP + d0;
      const uint4* gV = reinterpret_cast<const uint4*>(gv);
      vv.q[0] = gV[0]; vv.q[1] = gV[1];
#pragma unroll
      for (int e = 0; e < 16; ++e) Vs[(d0 + e) * 32 + key] = vv.u[e];
      if (k0 + 32 < AN) {  // prefetch next chunk (global_prefetch_b8)
        __builtin_prefetch(gk + (size_t)32 * kvs, 0, 3);
        __builtin_prefetch(gv + (size_t)32 * kvs, 0, 3);
      }
    }
#if defined(USE_ASYNC_LDS)
    __builtin_amdgcn_s_wait_asynccnt(0);
#endif
    __syncthreads();

    // S = Q K^T and online softmax for both row tiles
#pragma unroll
    for (int rt = 0; rt < 2; ++rt) {
      v8f s[2];
#pragma unroll
      for (int j = 0; j < 2; ++j) {
        v8f z = v8zero();
#pragma unroll
        for (int kc = 0; kc < 2; ++kc) {
          const u16* kp = &Ks[(j * 16 + lo) * 64 + kc * 32 + kb];
          z = wmma_bf16(aq[rt][kc], frag2(kp, kp + 16), z);
        }
        s[j] = z;
      }
#pragma unroll
      for (int r = 0; r < 8; ++r) {
        float cm = rmax16(fmaxf(s[0][r], s[1][r]));
        float mn = fmaxf(m[rt][r], cm);
        float sc = expf(m[rt][r] - mn);
        float p0 = expf(s[0][r] - mn);
        float p1 = expf(s[1][r] - mn);
        float rs = rsum16(p0 + p1);
        l[rt][r] = l[rt][r] * sc + rs;
        m[rt][r] = mn;
#pragma unroll
        for (int d = 0; d < 4; ++d) o[rt][d][r] *= sc;
        Pw[rt * 512 + (r + hi * 8) * 32 + lo]      = f2bf(p0);
        Pw[rt * 512 + (r + hi * 8) * 32 + 16 + lo] = f2bf(p1);
      }
    }
    __syncthreads();

    // O += P @ V ; V fragments register-reused across both row tiles
    v16bf bv[4];
#pragma unroll
    for (int d = 0; d < 4; ++d) {
      const u16* vp = &Vs[(d * 16 + lo) * 32 + kb];
      bv[d] = frag2(vp, vp + 16);
    }
#pragma unroll
    for (int rt = 0; rt < 2; ++rt) {
      const u16* pp = &Pw[rt * 512 + lo * 32 + kb];
      v16bf ap = frag2(pp, pp + 16);
#pragma unroll
      for (int d = 0; d < 4; ++d) o[rt][d] = wmma_bf16(ap, bv[d], o[rt][d]);
    }
  }

#pragma unroll
  for (int rt = 0; rt < 2; ++rt)
#pragma unroll
    for (int r = 0; r < 8; ++r) {
      float inv = 1.0f / l[rt][r];
      int row = q0 + wave * 32 + rt * 16 + r + hi * 8;
      float* op = Op + ((size_t)(b * AH + h) * AN + row) * ADP;
#pragma unroll
      for (int d = 0; d < 4; ++d) op[d * 16 + lo] = o[rt][d][r] * inv;
    }
}

// ---------------------------------------------------------------------------
// LayerNorm: one block per token row; emits fp32 and bf16
// ---------------------------------------------------------------------------
__global__ __launch_bounds__(256) void ln_kernel(const float* __restrict__ X,
                                                 const float* __restrict__ g,
                                                 const float* __restrict__ be,
                                                 float* __restrict__ Yf,
                                                 u16* __restrict__ Yb, int D) {
  const int row = blockIdx.x;
  const float* x = X + (size_t)row * D;
  float s = 0.0f, s2 = 0.0f;
  for (int c = threadIdx.x; c < D; c += 256) { float v = x[c]; s += v; s2 += v * v; }
#pragma unroll
  for (int msk = 16; msk >= 1; msk >>= 1) {
    s  += __shfl_xor(s, msk, 32);
    s2 += __shfl_xor(s2, msk, 32);
  }
  __shared__ float ws1[8], ws2[8];
  if ((threadIdx.x & 31) == 0) { ws1[threadIdx.x >> 5] = s; ws2[threadIdx.x >> 5] = s2; }
  __syncthreads();
  float ts = 0.0f, ts2 = 0.0f;
#pragma unroll
  for (int i = 0; i < 8; ++i) { ts += ws1[i]; ts2 += ws2[i]; }
  const float mu = ts / D;
  const float var = ts2 / D - mu * mu;
  const float rstd = rsqrtf(var + 1e-5f);
  for (int c = threadIdx.x; c < D; c += 256) {
    float v = (x[c] - mu) * rstd * g[c] + be[c];
    Yf[(size_t)row * D + c] = v;
    Yb[(size_t)row * D + c] = f2bf(v);
  }
}

__global__ __launch_bounds__(256) void add_kernel(const float* __restrict__ a,
                                                  const float* __restrict__ b,
                                                  float* __restrict__ c, int n4) {
  int i = blockIdx.x * 256 + threadIdx.x;
  if (i >= n4) return;
  float4 va = reinterpret_cast<const float4*>(a)[i];
  float4 vb = reinterpret_cast<const float4*>(b)[i];
  float4 vc; vc.x = va.x + vb.x; vc.y = va.y + vb.y; vc.z = va.z + vb.z; vc.w = va.w + vb.w;
  reinterpret_cast<float4*>(c)[i] = vc;
}

// ---------------------------------------------------------------------------
// Launcher
// ---------------------------------------------------------------------------
extern "C" void kernel_launch(void* const* d_in, const int* in_sizes, int n_in,
                              void* d_out, int out_size, void* d_ws, size_t ws_size,
                              hipStream_t stream) {
  (void)in_sizes; (void)n_in; (void)out_size; (void)ws_size;
  const float* x1    = (const float*)d_in[0];
  const float* x2    = (const float*)d_in[1];
  const float* Wq    = (const float*)d_in[2];
  const float* bq    = (const float*)d_in[3];
  const float* Wkv   = (const float*)d_in[4];
  const float* bkv   = (const float*)d_in[5];
  const float* Wqkv  = (const float*)d_in[6];
  const float* bqkv  = (const float*)d_in[7];
  const float* gamma = (const float*)d_in[8];
  const float* beta  = (const float*)d_in[9];
  const float* W1    = (const float*)d_in[10];
  const float* b1    = (const float*)d_in[11];
  const float* W2    = (const float*)d_in[12];
  const float* b2    = (const float*)d_in[13];
  float* out = (float*)d_out;

  constexpr int Bb = 4, Nn = 1024, Dd = 1024, Ff = 4096;
  constexpr int Mm = Bb * Nn;  // 4096 tokens

  char* w = (char*)d_ws;
  size_t off = 0;
  auto carve = [&](size_t bytes) -> void* {
    void* p = w + off;
    off = (off + bytes + 255) & ~(size_t)255;
    return p;
  };
  u16*   x1b  = (u16*)carve((size_t)Mm * Dd * 2);
  u16*   x2b  = (u16*)carve((size_t)Mm * Dd * 2);
  u16*   WqT  = (u16*)carve((size_t)Dd * Dd * 2);
  u16*   WkvT = (u16*)carve((size_t)Dd * 2 * Dd * 2);
  u16*   WqkvT= (u16*)carve((size_t)Dd * 3 * Dd * 2);
  u16*   W1T  = (u16*)carve((size_t)Dd * Ff * 2);
  u16*   W2T  = (u16*)carve((size_t)Ff * Dd * 2);
  u16*   qb   = (u16*)carve((size_t)Mm * Dd * 2);
  u16*   kvb  = (u16*)carve((size_t)Mm * 2 * Dd * 2);
  u16*   qkvb = (u16*)carve((size_t)Mm * 3 * Dd * 2);
  float* attn = (float*)carve((size_t)Mm * Dd * 4);
  float* xb   = (float*)carve((size_t)Mm * Dd * 4);
  float* nb   = (float*)carve((size_t)Mm * Dd * 4);
  u16*   nbf  = (u16*)carve((size_t)Mm * Dd * 2);
  u16*   hb   = (u16*)carve((size_t)Mm * Ff * 2);

  const int e4 = Mm * Dd / 4;  // 4M floats -> 1M float4

  // activations -> bf16
  cvt_bf16_kernel<<<e4 / 256, 256, 0, stream>>>(x1, x1b, e4);
  cvt_bf16_kernel<<<e4 / 256, 256, 0, stream>>>(x2, x2b, e4);
  // weights -> bf16 transposed [N][K]
  wtrans_bf16_kernel<<<(Dd * Dd) / 256, 256, 0, stream>>>(Wq, WqT, Dd, Dd);
  wtrans_bf16_kernel<<<(Dd * 2 * Dd) / 256, 256, 0, stream>>>(Wkv, WkvT, Dd, 2 * Dd);
  wtrans_bf16_kernel<<<(Dd * 3 * Dd) / 256, 256, 0, stream>>>(Wqkv, WqkvT, Dd, 3 * Dd);
  wtrans_bf16_kernel<<<(Dd * Ff) / 256, 256, 0, stream>>>(W1, W1T, Dd, Ff);
  wtrans_bf16_kernel<<<(Ff * Dd) / 256, 256, 0, stream>>>(W2, W2T, Ff, Dd);

  // q = x2 @ Wq + bq ; kv = x1 @ Wkv + bkv   (bf16 outputs for attention)
  gemm_bf16_kernel<<<dim3(Dd / 128, Mm / 128), 256, 0, stream>>>(
      x2b, WqT, bq, nullptr, nullptr, qb, Mm, Dd, Dd, 0);
  gemm_bf16_kernel<<<dim3(2 * Dd / 128, Mm / 128), 256, 0, stream>>>(
      x1b, WkvT, bkv, nullptr, nullptr, kvb, Mm, 2 * Dd, Dd, 0);

  // cross attention; flat (B,H,N,Dp) == reference reshape to (B,N,D)
  attn_kernel<<<dim3(Nn / 128, 16, Bb), 128, 0, stream>>>(
      qb, Dd, kvb, kvb + Dd, 2 * Dd, attn);

  // x = x2 + cross ; n = LN(x)
  add_kernel<<<e4 / 256, 256, 0, stream>>>(x2, attn, xb, e4);
  ln_kernel<<<Mm, 256, 0, stream>>>(xb, gamma, beta, nb, nbf, Dd);

  // qkv = n @ Wqkv + bqkv ; self attention ; x = n + attn2 ; n2 = LN(x)
  gemm_bf16_kernel<<<dim3(3 * Dd / 128, Mm / 128), 256, 0, stream>>>(
      nbf, WqkvT, bqkv, nullptr, nullptr, qkvb, Mm, 3 * Dd, Dd, 0);
  attn_kernel<<<dim3(Nn / 128, 16, Bb), 128, 0, stream>>>(
      qkvb, 3 * Dd, qkvb + Dd, qkvb + 2 * Dd, 3 * Dd, attn);
  add_kernel<<<e4 / 256, 256, 0, stream>>>(nb, attn, xb, e4);
  ln_kernel<<<Mm, 256, 0, stream>>>(xb, gamma, beta, nb, nbf, Dd);

  // h = gelu(n2 @ W1 + b1) ; out = n2 + h @ W2 + b2
  gemm_bf16_kernel<<<dim3(Ff / 128, Mm / 128), 256, 0, stream>>>(
      nbf, W1T, b1, nullptr, nullptr, hb, Mm, Ff, Dd, 1);
  gemm_bf16_kernel<<<dim3(Dd / 128, Mm / 128), 256, 0, stream>>>(
      hb, W2T, b2, nb, out, nullptr, Mm, Dd, Ff, 0);
}
